// LinearAttention_2113123910262
// MI455X (gfx1250) — matrix operand order repaired and checked by reading, not compiled, not run on hardware
//
#include <hip/hip_runtime.h>

// ---------------------------------------------------------------------------
// Softmax attention (B=2, C=64, N=4608) for MI455X / gfx1250, wave32.
//  * f16 operands + f32 accum on v_wmma_f32_16x16x32_f16 (8x K-depth of f32 WMMA)
//  * flash-style: the 2 x 4608^2 score matrix never leaves registers/LDS
//  * split-K x4 -> 2304 compute waves (parallelism is the scarce resource
//    for an ~11 GFLOP problem on this chip) + tiny combine kernel
//  * K/V tiles staged once per block (shared by 4 waves) in LDS via
//    global_load_async_to_lds_b128 (ASYNCcnt), TRIPLE buffered: DMA of tile
//    i+2 overlaps compute of tile i; bottom-of-loop wait is asynccnt<=4
//  * row-max via DPP row_ror butterfly (VALU, co-executes with XDL/WMMA)
//  * row-sum via a constant-ones B-tile in the PV WMMA (no reduction at all)
//  * 1/sqrt(C) folded into the Q projection
// ---------------------------------------------------------------------------

typedef _Float16 v16h __attribute__((ext_vector_type(16)));
typedef _Float16 v8h  __attribute__((ext_vector_type(8)));
typedef float    v8f  __attribute__((ext_vector_type(8)));

#define BSZ    2
#define CDIM   64
#define NTOK   4608
#define KSPLIT 4
#define NKEYS  (NTOK / KSPLIT)      // 1152 keys per split
#define NITER  (NKEYS / 32)         // 36 x 32-key steps
#define MTILES (NTOK / 16)          // 288 query tiles of 16 per batch
#define QKSCALE 0.125f              // 1/sqrt(64)

__device__ inline v16h cat8(v8h lo, v8h hi) {
    return __builtin_shufflevector(lo, hi, 0,1,2,3,4,5,6,7,8,9,10,11,12,13,14,15);
}

// max-butterfly over the 16-lane DPP row (lanes 0-15 / 16-31 independently)
template <int ROR>
__device__ inline float ror_max(float x) {
    int y = __builtin_amdgcn_update_dpp(0, __float_as_int(x),
                                        0x120 + ROR, 0xF, 0xF, true);
    return fmaxf(x, __int_as_float(y));
}
__device__ inline float rowmax16(float x) {
    x = ror_max<1>(x); x = ror_max<2>(x);
    x = ror_max<4>(x); x = ror_max<8>(x);
    return x;   // all 16 lanes of the row group now hold the max
}

// CDNA5 async DMA: 16 bytes global -> LDS, tracked by ASYNCcnt
__device__ inline void async_cp16(unsigned lds_off, unsigned long long saddr,
                                  unsigned goff) {
    asm volatile("global_load_async_to_lds_b128 %0, %1, %2"
                 :: "v"(lds_off), "v"(goff), "s"(saddr) : "memory");
}

// ---------------------------------------------------------------------------
// Kernel 1: 1x1-conv projections -> f16 in WMMA-friendly layouts
//   q_t,k_t : token-major [B][N][64]   v_cm : channel-major [B][64][N]
//   q is pre-scaled by 1/sqrt(C) so score WMMA output feeds softmax directly
// ---------------------------------------------------------------------------
__global__ __launch_bounds__(256) void proj_qkv(
    const float* __restrict__ x,  const float* __restrict__ h,
    const float* __restrict__ Wq, const float* __restrict__ bq,
    const float* __restrict__ Wk, const float* __restrict__ bk,
    const float* __restrict__ Wv, const float* __restrict__ bv,
    _Float16* __restrict__ q_t, _Float16* __restrict__ k_t,
    _Float16* __restrict__ v_cm)
{
    __shared__ float xs[64][65];
    __shared__ float hs[64][65];

    const int b   = blockIdx.x / (NTOK/64);
    const int t0  = (blockIdx.x % (NTOK/64)) * 64;
    const int tid = threadIdx.x;

    for (int i = 0; i < 16; ++i) {
        int id = tid + i*256;
        int c = id >> 6, j = id & 63;
        xs[c][j] = x[(size_t)(b*CDIM + c)*NTOK + t0 + j];
        hs[c][j] = h[(size_t)(b*CDIM + c)*NTOK + t0 + j];
    }
    __syncthreads();

    {   // q,k: fixed o per thread -> stores coalesced over o (token-major)
        const int o  = tid & 63;
        const int jb = tid >> 6;
        const float bqo = bq[o], bko = bk[o];
        for (int i = 0; i < 16; ++i) {
            const int j = jb*16 + i;
            float accq = bqo, acck = bko;
            #pragma unroll
            for (int c = 0; c < 64; ++c) {
                accq += Wq[o*64 + c] * xs[c][j];
                acck += Wk[o*64 + c] * hs[c][j];
            }
            q_t[(size_t)(b*NTOK + t0 + j)*64 + o] = (_Float16)(accq * QKSCALE);
            k_t[(size_t)(b*NTOK + t0 + j)*64 + o] = (_Float16)acck;
        }
    }
    {   // v: fixed token per thread -> stores coalesced over j, Wv row scalar
        const int j  = tid & 63;
        const int ob = tid >> 6;
        for (int i = 0; i < 16; ++i) {
            const int o = ob*16 + i;
            float acc = bv[o];
            #pragma unroll
            for (int c = 0; c < 64; ++c)
                acc += Wv[o*64 + c] * hs[c][j];
            v_cm[(size_t)(b*CDIM + o)*NTOK + t0 + j] = (_Float16)acc;
        }
    }
}

// ---------------------------------------------------------------------------
// Kernel 2: flash attention partials.
// Block = 4 waves = one 64-query tile x one key split (grid 576).
// Wave owns 16 queries; waves share async-staged K/V tiles in LDS.
// ---------------------------------------------------------------------------
__global__ __launch_bounds__(128) void flash_attn(
    const _Float16* __restrict__ q_t,
    const _Float16* __restrict__ k_t,
    const _Float16* __restrict__ v_cm,
    float* __restrict__ part_O,      // [B*288][KSPLIT][16][64]
    float* __restrict__ part_m,      // [B*288][KSPLIT][16]
    float* __restrict__ part_l)      // [B*288][KSPLIT][16]
{
    // bank-padded, triple-buffered staging
    // (row strides 36 / 20 dwords; gcd(.,64)=4 -> 16 distinct banks)
    __shared__ __attribute__((aligned(16))) _Float16 kbuf[3][32][72];
    __shared__ __attribute__((aligned(16))) _Float16 vbuf[3][64][40];
    __shared__ __attribute__((aligned(16))) _Float16 pbuf[4][16][40];

    const int tid   = threadIdx.x;
    const int wave  = tid >> 5;
    const int lane  = tid & 31;
    const int half  = lane >> 4;
    const int l16   = lane & 15;

    const int split = blockIdx.x & (KSPLIT - 1);
    const int m64   = (blockIdx.x >> 2) % (NTOK/64);
    const int b     = (blockIdx.x >> 2) / (NTOK/64);
    const int m0    = m64*64 + wave*16;
    const int tile  = b*MTILES + m64*4 + wave;     // 16-query tile id
    const int nbase = split * NKEYS;

    const _Float16* qb = q_t  + (size_t)b*NTOK*CDIM;
    const unsigned long long kb = (unsigned long long)(k_t  + (size_t)b*NTOK*CDIM);
    const unsigned long long vb = (unsigned long long)(v_cm + (size_t)b*CDIM*NTOK);

    // branchless DMA map: every thread moves 2 K-chunks + 2 V-chunks (16B each)
    // per 32-key tile (512 x 16B total), no EXEC juggling around the WMMAs.
    auto issue_tile = [&](int nb, int n0) {
        #pragma unroll
        for (int i = 0; i < 2; ++i) {
            const int ch = tid + i*128;                        // 0..255
            const int krow = ch >> 3, kcol = ch & 7;
            async_cp16((unsigned)(size_t)&kbuf[nb][krow][kcol*8], kb,
                       (unsigned)(((n0 + krow)*64 + kcol*8) * 2));
            const int vc = ch >> 2, vcol = ch & 3;
            async_cp16((unsigned)(size_t)&vbuf[nb][vc][vcol*8], vb,
                       (unsigned)((vc*NTOK + n0 + vcol*8) * 2));
        }
    };

    // preload Q A-fragments (lane: row M=l16; halves {half*8..}+{16+half*8..})
    const _Float16* qrow = qb + (size_t)(m0 + l16)*64;
    const v16h aq0 = cat8(*(const v8h*)(qrow      + half*8),
                          *(const v8h*)(qrow + 16 + half*8));
    const v16h aq1 = cat8(*(const v8h*)(qrow + 32 + half*8),
                          *(const v8h*)(qrow + 48 + half*8));

    v16h bones;                       // constant-ones B-tile: row-sum via WMMA
    #pragma unroll
    for (int i = 0; i < 16; ++i) bones[i] = (_Float16)1.0f;

    float mrun[8];
    v8f oacc[5];                      // 4 c-tiles + ones-tile (= running l)
    #pragma unroll
    for (int r = 0; r < 8; ++r) mrun[r] = -1e30f;
    #pragma unroll
    for (int t = 0; t < 5; ++t) oacc[t] = v8f{};

    // prologue: DMA tiles 0 and 1, wait for tile 0 only (4 oldest ops)
    issue_tile(0, nbase);
    issue_tile(1, nbase + 32);
    asm volatile("s_wait_asynccnt 0x4" ::: "memory");
    __syncthreads();

    for (int it = 0; it < NITER; ++it) {
        const int cur = it % 3;
        const bool more = (it + 2 < NITER);
        if (more)                                   // DMA tile it+2 in flight
            issue_tile((it + 2) % 3, nbase + (it + 2)*32);

        // ---- K B-fragments: all four first, then interleaved WMMAs ----
        const _Float16* k0 = &kbuf[cur][l16][0];
        const _Float16* k1 = &kbuf[cur][16 + l16][0];
        const v16h b00 = cat8(*(const v8h*)(k0      + half*8),
                              *(const v8h*)(k0      + half*8 + 16));
        const v16h b01 = cat8(*(const v8h*)(k0 + 32 + half*8),
                              *(const v8h*)(k0 + 32 + half*8 + 16));
        const v16h b10 = cat8(*(const v8h*)(k1      + half*8),
                              *(const v8h*)(k1      + half*8 + 16));
        const v16h b11 = cat8(*(const v8h*)(k1 + 32 + half*8),
                              *(const v8h*)(k1 + 32 + half*8 + 16));
        v8f s0 = v8f{}, s1 = v8f{};
        s0 = __builtin_amdgcn_wmma_f32_16x16x32_f16(false, aq0, false, b00,
                                                    (short)0, s0, false, false);
        s1 = __builtin_amdgcn_wmma_f32_16x16x32_f16(false, aq0, false, b10,
                                                    (short)0, s1, false, false);
        s0 = __builtin_amdgcn_wmma_f32_16x16x32_f16(false, aq1, false, b01,
                                                    (short)0, s0, false, false);
        s1 = __builtin_amdgcn_wmma_f32_16x16x32_f16(false, aq1, false, b11,
                                                    (short)0, s1, false, false);

        // ---- V B-fragments issued now; latency hides under softmax VALU ----
        v16h bv4[4];
        #pragma unroll
        for (int t = 0; t < 4; ++t) {
            const _Float16* vrow = &vbuf[cur][t*16 + l16][half*8];
            bv4[t] = cat8(*(const v8h*)(vrow), *(const v8h*)(vrow + 16));
        }

        // ---- online softmax; row m = half*8+r, col = l16 ----
        #pragma unroll
        for (int r = 0; r < 8; ++r) {
            const float mnew = fmaxf(mrun[r], rowmax16(fmaxf(s0[r], s1[r])));
            const float corr = __expf(mrun[r] - mnew);
            const float p0   = __expf(s0[r] - mnew);
            const float p1   = __expf(s1[r] - mnew);
            mrun[r] = mnew;
            #pragma unroll
            for (int t = 0; t < 5; ++t) oacc[t][r] *= corr;
            const int mrow = half*8 + r;
            pbuf[wave][mrow][l16]      = (_Float16)p0;
            pbuf[wave][mrow][l16 + 16] = (_Float16)p1;
        }
        asm volatile("s_wait_dscnt 0" ::: "memory");   // D->A transpose of P

        const _Float16* prow = &pbuf[wave][l16][0];
        const v16h pA = cat8(*(const v8h*)(prow      + half*8),
                             *(const v8h*)(prow + 16 + half*8));

        // ---- O += P V^T (4 c-tiles) and l += P . 1 (ones-tile) ----
        #pragma unroll
        for (int t = 0; t < 4; ++t)
            oacc[t] = __builtin_amdgcn_wmma_f32_16x16x32_f16(false, pA, false,
                                                             bv4[t], (short)0,
                                                             oacc[t], false, false);
        oacc[4] = __builtin_amdgcn_wmma_f32_16x16x32_f16(false, pA, false, bones,
                                                         (short)0, oacc[4],
                                                         false, false);

        // tile it+1 must have landed before the next iteration reads it;
        // the 4 ops just issued for tile it+2 may stay in flight.
        if (more) asm volatile("s_wait_asynccnt 0x4" ::: "memory");
        else      asm volatile("s_wait_asynccnt 0x0" ::: "memory");
        __syncthreads();
    }

    // ---- spill split partials ----
    const size_t pb = ((size_t)tile*KSPLIT + split) * 16;
    #pragma unroll
    for (int r = 0; r < 8; ++r) {
        const int mrow = half*8 + r;
        #pragma unroll
        for (int t = 0; t < 4; ++t)
            part_O[(pb + mrow)*64 + t*16 + l16] = oacc[t][r];
        if (l16 == 0) {
            part_m[pb + mrow] = mrun[r];
            part_l[pb + mrow] = oacc[4][r];   // ones-tile col == row sum
        }
    }
}

// ---------------------------------------------------------------------------
// Kernel 3: merge the KSPLIT partials, normalize, write out[b][c][m] (f32)
// ---------------------------------------------------------------------------
__global__ __launch_bounds__(128) void combine(
    const float* __restrict__ part_O,
    const float* __restrict__ part_m,
    const float* __restrict__ part_l,
    float* __restrict__ out)
{
    const int tile = blockIdx.x;                 // b*288 + mt16
    const int b    = tile / MTILES;
    const int m0   = (tile % MTILES) * 16;
    const int ml   = threadIdx.x & 15;
    const int cb   = threadIdx.x >> 4;           // 0..7
    const size_t pb = (size_t)tile * KSPLIT * 16;

    float mm[KSPLIT], M = -1e30f;
    #pragma unroll
    for (int s = 0; s < KSPLIT; ++s) {
        mm[s] = part_m[pb + s*16 + ml];
        M = fmaxf(M, mm[s]);
    }
    float w[KSPLIT], denom = 0.0f;
    #pragma unroll
    for (int s = 0; s < KSPLIT; ++s) {
        w[s] = __expf(mm[s] - M);
        denom += w[s] * part_l[pb + s*16 + ml];
    }
    const float inv = 1.0f / denom;

    for (int i = 0; i < 8; ++i) {
        const int c = cb*8 + i;
        float acc = 0.0f;
        #pragma unroll
        for (int s = 0; s < KSPLIT; ++s)
            acc += w[s] * part_O[(pb + s*16 + ml)*64 + c];
        out[((size_t)b*CDIM + c)*NTOK + m0 + ml] = acc * inv;
    }
}

// ---------------------------------------------------------------------------
extern "C" void kernel_launch(void* const* d_in, const int* in_sizes, int n_in,
                              void* d_out, int out_size, void* d_ws, size_t ws_size,
                              hipStream_t stream)
{
    const float* x  = (const float*)d_in[0];
    const float* h  = (const float*)d_in[1];
    const float* Wq = (const float*)d_in[2];
    const float* bq = (const float*)d_in[3];
    const float* Wk = (const float*)d_in[4];
    const float* bk = (const float*)d_in[5];
    const float* Wv = (const float*)d_in[6];
    const float* bv = (const float*)d_in[7];
    float* out = (float*)d_out;

    // workspace carve-up (16B-aligned sections)
    const size_t qkv = (size_t)BSZ*NTOK*CDIM;          // halves per tensor
    _Float16* q_t  = (_Float16*)d_ws;
    _Float16* k_t  = q_t + qkv;
    _Float16* v_cm = k_t + qkv;
    float* part_O = (float*)(v_cm + qkv);              // 576*4*16*64 f32
    float* part_m = part_O + (size_t)BSZ*MTILES*KSPLIT*16*64;
    float* part_l = part_m + (size_t)BSZ*MTILES*KSPLIT*16;

    proj_qkv<<<BSZ*(NTOK/64), 256, 0, stream>>>(x, h, Wq, bq, Wk, bk, Wv, bv,
                                                q_t, k_t, v_cm);
    flash_attn<<<BSZ*(NTOK/64)*KSPLIT, 128, 0, stream>>>(q_t, k_t, v_cm,
                                                         part_O, part_m, part_l);
    combine<<<BSZ*MTILES, 128, 0, stream>>>(part_O, part_m, part_l, out);
}
